// SA_v_26920855011331
// MI455X (gfx1250) — matrix operand order repaired and checked by reading, not compiled, not run on hardware
//
#include <hip/hip_runtime.h>
#include <hip/hip_bf16.h>
#include <stdint.h>

// Problem constants (from reference)
#define B_  32
#define G_  32
#define S_  1024
#define D_  512
#define H_  16
#define HD_ 32
#define SCALE_Q 0.17677669529663687f   // 1/sqrt(HD)

typedef __attribute__((ext_vector_type(16))) __bf16 v16bf;
typedef __attribute__((ext_vector_type(8)))  float  v8f;
typedef __attribute__((ext_vector_type(4)))  unsigned int v4u;
typedef __attribute__((ext_vector_type(4)))  int v4i;
typedef __attribute__((ext_vector_type(8)))  int v8i;

union BF8 { uint4 u; __bf16 h[8]; };

// inv_freq[j] = 10000^(-j/8) = 10^(-j/2)
__constant__ float c_invf[8] = {
    1.0f, 0.31622776601683794f, 0.1f, 0.031622776601683794f,
    0.01f, 0.0031622776601683794f, 0.001f, 0.00031622776601683794f
};

// ---------------------------------------------------------------------------
// TDM: one descriptor moves a contiguous 16x512 f32 tile (32 KB) into LDS.
// D# per CDNA5 ISA 8.3/8.4: group0 = {count/type/addr}, group1 = {dims/strides}
// ---------------------------------------------------------------------------
__device__ __forceinline__ void tdm_load_tile_16x512_f32(const float* gsrc,
                                                         unsigned lds_addr)
{
    unsigned long long ga = (unsigned long long)(uintptr_t)gsrc;
    v4u g0;
    g0[0] = 1u;                                        // count=1 (valid user D#)
    g0[1] = lds_addr;                                  // lds_addr [63:32]
    g0[2] = (unsigned)(ga & 0xFFFFFFFFu);              // global_addr lo
    g0[3] = (unsigned)((ga >> 32) & 0x01FFFFFFu)       // global_addr hi
          | 0x80000000u;                               // type=2 ("image")
    v8i g1;
    g1[0] = 0x20000;               // workgroup_mask=0, data_size=2 (4 bytes)
    g1[1] = (int)(512u << 16);     // tensor_dim0 = 512 (bits 79:48, low part)
    g1[2] = (int)(16u << 16);      // tensor_dim0 hi=0 ; tensor_dim1 = 16
    g1[3] = (int)(512u << 16);     // tensor_dim1 hi=0 ; tile_dim0 = 512
    g1[4] = 16;                    // tile_dim1 = 16, tile_dim2 = 0
    g1[5] = 512;                   // tensor_dim0_stride = 512 (lo 32)
    g1[6] = 0;                     // stride hi, tensor_dim1_stride lo
    g1[7] = 0;
    v4i z4 = {0, 0, 0, 0};
#if defined(__clang_major__) && (__clang_major__ >= 23)
    v8i z8 = {0, 0, 0, 0, 0, 0, 0, 0};
    __builtin_amdgcn_tensor_load_to_lds(g0, g1, z4, z4, z8, 0);
#else
    __builtin_amdgcn_tensor_load_to_lds(g0, g1, z4, z4, 0);
#endif
}

// ---------------------------------------------------------------------------
// Kernel 0: weight conversion f32 -> bf16 (scale folded into Wq)
// ---------------------------------------------------------------------------
__global__ __launch_bounds__(256)
void prep_kernel(const float* __restrict__ Wq, const float* __restrict__ Wk,
                 const float* __restrict__ Wv, const float* __restrict__ Wm,
                 __bf16* __restrict__ Wq_bf, __bf16* __restrict__ Wk_bf,
                 __bf16* __restrict__ Wv_bf, __bf16* __restrict__ Wm_bf)
{
    int i = blockIdx.x * 256 + threadIdx.x;   // grid covers exactly 512*512
    Wq_bf[i] = (__bf16)(Wq[i] * SCALE_Q);
    Wk_bf[i] = (__bf16)(Wk[i]);
    Wv_bf[i] = (__bf16)(Wv[i]);
    Wm_bf[i] = (__bf16)(Wm[i]);
}

// ---------------------------------------------------------------------------
// Kernel 1: QKV projection (+bias, +RoPE for Q/K), outputs bf16
//   grid = (64 row-tiles, 3 matrices, B), block = 256 (8 waves)
//   x tile staged into LDS by the Tensor Data Mover (wave 0 issues).
//   Q,K layout: [B][H][S][HD]   V layout: [B][H][HD][S] (feature-major)
// ---------------------------------------------------------------------------
__global__ __launch_bounds__(256)
void qkv_rope_kernel(const float* __restrict__ x,
                     const __bf16* __restrict__ Wq_bf,
                     const __bf16* __restrict__ Wk_bf,
                     const __bf16* __restrict__ Wv_bf,
                     const float* __restrict__ bq, const float* __restrict__ bk,
                     __bf16* __restrict__ qb, __bf16* __restrict__ kb,
                     __bf16* __restrict__ vTb)
{
    const int tid  = threadIdx.x;
    const int lane = tid & 31;
    const int wv   = tid >> 5;          // wave id 0..7
    const int mt   = blockIdx.x;        // row tile 0..63
    const int sel  = blockIdx.y;        // 0=Q 1=K 2=V
    const int b    = blockIdx.z;
    const int s0   = mt * 16;
    const int hi   = (lane >= 16) ? 1 : 0;
    const int l15  = lane & 15;

    __shared__ __align__(16) float smem[16 * 512];   // 32 KB x-tile (f32)

    // ---- TDM DMA of the contiguous 16x512 f32 tile into LDS ---------------
    if (wv == 0) {
        tdm_load_tile_16x512_f32(x + ((size_t)b * S_ + s0) * D_,
                                 (unsigned)(uintptr_t)(&smem[0]));
        __builtin_amdgcn_s_wait_tensorcnt(0);
    }
    __syncthreads();

    const __bf16* Wsel = (sel == 0) ? Wq_bf : (sel == 1) ? Wk_bf : Wv_bf;

    v8f acc[4] = {};

    // ---- K loop: D = A(16x32 bf16, from LDS f32) x B(32x16 bf16, weights) --
    for (int k0 = 0; k0 < 512; k0 += 32) {
        v16bf a;
        {
            // A layout: lane m(0..15): row m, K = 0..15;
            //           lane m+16   : row m, K = 16..31
            const float* row = &smem[l15 * 512 + k0 + (hi ? 16 : 0)];
            float4 f0 = *(const float4*)(row);
            float4 f1 = *(const float4*)(row + 4);
            float4 f2 = *(const float4*)(row + 8);
            float4 f3 = *(const float4*)(row + 12);
            a[0] = (__bf16)f0.x;  a[1] = (__bf16)f0.y;
            a[2] = (__bf16)f0.z;  a[3] = (__bf16)f0.w;
            a[4] = (__bf16)f1.x;  a[5] = (__bf16)f1.y;
            a[6] = (__bf16)f1.z;  a[7] = (__bf16)f1.w;
            a[8] = (__bf16)f2.x;  a[9] = (__bf16)f2.y;
            a[10] = (__bf16)f2.z; a[11] = (__bf16)f2.w;
            a[12] = (__bf16)f3.x; a[13] = (__bf16)f3.y;
            a[14] = (__bf16)f3.z; a[15] = (__bf16)f3.w;
        }
#pragma unroll
        for (int t = 0; t < 4; ++t) {
            // B layout: lane n(0..15): K=0..15 of column n; lane n+16: K=16..31
            int n = wv * 64 + t * 16 + l15;
            const __bf16* wrow = Wsel + (size_t)n * 512 + k0 + (hi ? 16 : 0);
            BF8 c0, c1;
            c0.u = *(const uint4*)(wrow);
            c1.u = *(const uint4*)(wrow + 8);
            v16bf bf;
#pragma unroll
            for (int i = 0; i < 8; ++i) { bf[i] = c0.h[i]; bf[8 + i] = c1.h[i]; }
            acc[t] = __builtin_amdgcn_wmma_f32_16x16x32_bf16(
                false, a, false, bf, (short)0, acc[t], false, false);
        }
    }

    // ---- epilogue: bias, RoPE (Q/K), convert + scatter store --------------
    const float invf = c_invf[l15 & 7];
#pragma unroll
    for (int t = 0; t < 4; ++t) {
        int n    = wv * 64 + t * 16 + l15;   // global feature 0..511
        int head = n >> 5;
        int xh   = (n >> 4) & 1;             // 0 = y-half, 1 = x-half
        int i16  = n & 15;
        float bias = 0.0f;
        if (sel == 0)      bias = bq[n] * SCALE_Q;
        else if (sel == 1) bias = bk[n];
#pragma unroll
        for (int r = 0; r < 8; ++r) {
            int M = r + (hi ? 8 : 0);
            int s = s0 + M;                  // token index
            float val = acc[t][r] + bias;
            if (sel < 2) {
                float pos = (float)(xh ? (s & 31) : (s >> 5));  // px : py
                float ang = pos * invf;
                float cv = __cosf(ang), sv = __sinf(ang);
                float partner = __shfl_xor(val, 8, 32);
                float rot = (i16 < 8) ? -partner : partner;
                val = val * cv + rot * sv;
            }
            __bf16 o = (__bf16)val;
            if (sel == 0)
                qb[(((size_t)b * H_ + head) * S_ + s) * HD_ + (n & 31)] = o;
            else if (sel == 1)
                kb[(((size_t)b * H_ + head) * S_ + s) * HD_ + (n & 31)] = o;
            else
                vTb[(((size_t)b * H_ + head) * HD_ + (n & 31)) * S_ + s] = o;
        }
    }
}

// ---------------------------------------------------------------------------
// Kernel 2: flash attention, one wave per (b, h, 16-query tile).
// Processes TWO key tiles per iteration so the ctx WMMA uses its full K=32
// and all 32 lanes carry V data.
// ---------------------------------------------------------------------------
__global__ __launch_bounds__(32)
void attn_kernel(const __bf16* __restrict__ qb, const __bf16* __restrict__ kb,
                 const __bf16* __restrict__ vTb, __bf16* __restrict__ ctx)
{
    const int lane = threadIdx.x & 31;
    const int qt   = blockIdx.x;     // 0..63
    const int h    = blockIdx.y;
    const int b    = blockIdx.z;
    const int hi   = (lane >= 16) ? 1 : 0;
    const int l15  = lane & 15;
    const size_t hb = (size_t)b * H_ + h;

    __shared__ __align__(16) __bf16 lds_p[16][32];

    // Q A-frag (16 queries x 32 feats), already scaled by 1/sqrt(HD)
    v16bf aq;
    {
        const __bf16* qrow = qb + (hb * S_ + qt * 16 + l15) * HD_ + (hi ? 16 : 0);
        BF8 c0, c1;
        c0.u = *(const uint4*)(qrow);
        c1.u = *(const uint4*)(qrow + 8);
#pragma unroll
        for (int i = 0; i < 8; ++i) { aq[i] = c0.h[i]; aq[8 + i] = c1.h[i]; }
    }

    float mrow[8], lrow[8];
#pragma unroll
    for (int r = 0; r < 8; ++r) { mrow[r] = -1e30f; lrow[r] = 0.0f; }
    v8f acc0 = {}, acc1 = {};

    for (int kt2 = 0; kt2 < 32; ++kt2) {       // 32 iterations x 32 keys
        const int k32 = kt2 * 32;
        // two K B-frags (32 feats x 16 key tokens each)
        v16bf bk0, bk1;
        {
            const __bf16* kr0 = kb + (hb * S_ + k32 + l15) * HD_ + (hi ? 16 : 0);
            const __bf16* kr1 = kr0 + 16 * HD_;
            BF8 c0, c1, c2, c3;
            c0.u = *(const uint4*)(kr0);
            c1.u = *(const uint4*)(kr0 + 8);
            c2.u = *(const uint4*)(kr1);
            c3.u = *(const uint4*)(kr1 + 8);
#pragma unroll
            for (int i = 0; i < 8; ++i) {
                bk0[i] = c0.h[i]; bk0[8 + i] = c1.h[i];
                bk1[i] = c2.h[i]; bk1[8 + i] = c3.h[i];
            }
        }
        v8f z0 = {}, z1 = {};
        v8f sc0 = __builtin_amdgcn_wmma_f32_16x16x32_bf16(
            false, aq, false, bk0, (short)0, z0, false, false);
        v8f sc1 = __builtin_amdgcn_wmma_f32_16x16x32_bf16(
            false, aq, false, bk1, (short)0, z1, false, false);

        // online softmax per row over the 32 fresh keys
        float pr0[8], pr1[8];
#pragma unroll
        for (int r = 0; r < 8; ++r) {
            float v0 = sc0[r], v1 = sc1[r];
            float mx = fmaxf(v0, v1);
            mx = fmaxf(mx, __shfl_xor(mx, 1, 32));
            mx = fmaxf(mx, __shfl_xor(mx, 2, 32));
            mx = fmaxf(mx, __shfl_xor(mx, 4, 32));
            mx = fmaxf(mx, __shfl_xor(mx, 8, 32));
            float mnew  = fmaxf(mrow[r], mx);
            float alpha = __expf(mrow[r] - mnew);
            float p0    = __expf(v0 - mnew);
            float p1    = __expf(v1 - mnew);
            float ps = p0 + p1;
            ps += __shfl_xor(ps, 1, 32);
            ps += __shfl_xor(ps, 2, 32);
            ps += __shfl_xor(ps, 4, 32);
            ps += __shfl_xor(ps, 8, 32);
            lrow[r] = lrow[r] * alpha + ps;
            mrow[r] = mnew;
            acc0[r] *= alpha;
            acc1[r] *= alpha;
            pr0[r] = p0;
            pr1[r] = p1;
        }

        // C-layout -> A-layout transpose of P(16x32) through LDS
        __syncthreads();
#pragma unroll
        for (int r = 0; r < 8; ++r) {
            int M = r + (hi ? 8 : 0);
            lds_p[M][l15]      = (__bf16)pr0[r];
            lds_p[M][16 + l15] = (__bf16)pr1[r];
        }
        __syncthreads();

        v16bf ap;
        {
            BF8 c0, c1;
            c0.u = *(const uint4*)(&lds_p[l15][hi ? 16 : 0]);
            c1.u = *(const uint4*)(&lds_p[l15][8 + (hi ? 16 : 0)]);
#pragma unroll
            for (int i = 0; i < 8; ++i) { ap[i] = c0.h[i]; ap[8 + i] = c1.h[i]; }
        }

        // ctx += P(16x32) x V(32 keys x 16 feats), two N tiles; lanes lo carry
        // keys 0..15, lanes hi carry keys 16..31 of this 32-key group.
#pragma unroll
        for (int t = 0; t < 2; ++t) {
            const __bf16* vrow =
                vTb + (hb * HD_ + t * 16 + l15) * S_ + k32 + (hi ? 16 : 0);
            BF8 c0, c1;
            c0.u = *(const uint4*)(vrow);
            c1.u = *(const uint4*)(vrow + 8);
            v16bf bv;
#pragma unroll
            for (int i = 0; i < 8; ++i) { bv[i] = c0.h[i]; bv[8 + i] = c1.h[i]; }
            if (t == 0)
                acc0 = __builtin_amdgcn_wmma_f32_16x16x32_bf16(
                    false, ap, false, bv, (short)0, acc0, false, false);
            else
                acc1 = __builtin_amdgcn_wmma_f32_16x16x32_bf16(
                    false, ap, false, bv, (short)0, acc1, false, false);
        }
    }

    // normalize and store merged ctx [B][S][D]
#pragma unroll
    for (int r = 0; r < 8; ++r) {
        int M = r + (hi ? 8 : 0);
        float inv = 1.0f / lrow[r];
        size_t base = ((size_t)b * S_ + qt * 16 + M) * D_ + h * HD_;
        ctx[base + l15]      = (__bf16)(acc0[r] * inv);
        ctx[base + 16 + l15] = (__bf16)(acc1[r] * inv);
    }
}

// ---------------------------------------------------------------------------
// Kernel 3: output projection + residual + LayerNorm
//   grid = (64 row-tiles, B), block = 256 (8 waves)
//   Residual x-tile prefetched to LDS asynchronously, awaited after the GEMM.
// ---------------------------------------------------------------------------
__global__ __launch_bounds__(256)
void proj_ln_kernel(const __bf16* __restrict__ ctx, const __bf16* __restrict__ Wm_bf,
                    const float* __restrict__ bm, const float* __restrict__ x,
                    const float* __restrict__ gamma, const float* __restrict__ beta,
                    float* __restrict__ out)
{
    const int tid  = threadIdx.x;
    const int lane = tid & 31;
    const int wv   = tid >> 5;
    const int mt   = blockIdx.x;
    const int b    = blockIdx.y;
    const int s0   = mt * 16;
    const int hi   = (lane >= 16) ? 1 : 0;
    const int l15  = lane & 15;

    __shared__ __align__(16) float xres[16 * 512];   // 32 KB residual tile
    __shared__ float hsum[16];
    __shared__ float hsq[16];

    // ---- kick off async copy of the residual x tile (overlaps the GEMM) ---
    {
        const float4* gsrc = (const float4*)(x + ((size_t)b * S_ + s0) * D_);
        unsigned lds_base = (unsigned)(uintptr_t)(&xres[0]);
#pragma unroll
        for (int i = 0; i < 8; ++i) {
            int e = tid + i * 256;                    // float4 index, 2048 total
            unsigned loff = lds_base + e * 16;
            const float4* g = gsrc + e;
            asm volatile("global_load_async_to_lds_b128 %0, %1, off"
                         :: "v"(loff), "v"(g) : "memory");
        }
    }
    if (tid < 16) { hsum[tid] = 0.0f; hsq[tid] = 0.0f; }

    v8f acc[4] = {};
    const __bf16* arow_base = ctx + ((size_t)b * S_ + s0 + l15) * D_;

    for (int k0 = 0; k0 < 512; k0 += 32) {
        v16bf a;
        {
            const __bf16* ar = arow_base + k0 + (hi ? 16 : 0);
            BF8 c0, c1;
            c0.u = *(const uint4*)(ar);
            c1.u = *(const uint4*)(ar + 8);
#pragma unroll
            for (int i = 0; i < 8; ++i) { a[i] = c0.h[i]; a[8 + i] = c1.h[i]; }
        }
#pragma unroll
        for (int t = 0; t < 4; ++t) {
            int n = wv * 64 + t * 16 + l15;
            const __bf16* wrow = Wm_bf + (size_t)n * 512 + k0 + (hi ? 16 : 0);
            BF8 c0, c1;
            c0.u = *(const uint4*)(wrow);
            c1.u = *(const uint4*)(wrow + 8);
            v16bf bf;
#pragma unroll
            for (int i = 0; i < 8; ++i) { bf[i] = c0.h[i]; bf[8 + i] = c1.h[i]; }
            acc[t] = __builtin_amdgcn_wmma_f32_16x16x32_bf16(
                false, a, false, bf, (short)0, acc[t], false, false);
        }
    }

    // residual tile must be in LDS now
    asm volatile("s_wait_asynccnt 0" ::: "memory");
    __syncthreads();

    // h = proj + bias + residual, accumulate LN stats
    float hv[4][8];
#pragma unroll
    for (int t = 0; t < 4; ++t) {
        int n = wv * 64 + t * 16 + l15;
        float bmv = bm[n];
#pragma unroll
        for (int r = 0; r < 8; ++r) {
            int M = r + (hi ? 8 : 0);
            hv[t][r] = acc[t][r] + bmv + xres[M * 512 + n];
        }
    }
#pragma unroll
    for (int r = 0; r < 8; ++r) {
        int M = r + (hi ? 8 : 0);
        float ps = hv[0][r] + hv[1][r] + hv[2][r] + hv[3][r];
        float pq = hv[0][r] * hv[0][r] + hv[1][r] * hv[1][r] +
                   hv[2][r] * hv[2][r] + hv[3][r] * hv[3][r];
        atomicAdd(&hsum[M], ps);
        atomicAdd(&hsq[M], pq);
    }
    __syncthreads();

#pragma unroll
    for (int t = 0; t < 4; ++t) {
        int n = wv * 64 + t * 16 + l15;
        float g = gamma[n], be = beta[n];
#pragma unroll
        for (int r = 0; r < 8; ++r) {
            int M = r + (hi ? 8 : 0);
            int s = s0 + M;
            float mu  = hsum[M] * (1.0f / 512.0f);
            float var = hsq[M] * (1.0f / 512.0f) - mu * mu;
            float rs  = rsqrtf(var + 1e-5f);
            out[((size_t)b * S_ + s) * D_ + n] = (hv[t][r] - mu) * rs * g + be;
        }
    }
}

// ---------------------------------------------------------------------------
extern "C" void kernel_launch(void* const* d_in, const int* in_sizes, int n_in,
                              void* d_out, int out_size, void* d_ws, size_t ws_size,
                              hipStream_t stream)
{
    const float* x     = (const float*)d_in[0];
    const float* Wv    = (const float*)d_in[1];
    const float* Wk    = (const float*)d_in[2];
    const float* bk    = (const float*)d_in[3];
    const float* Wq    = (const float*)d_in[4];
    const float* bq    = (const float*)d_in[5];
    const float* Wm    = (const float*)d_in[6];
    const float* bm    = (const float*)d_in[7];
    const float* gamma = (const float*)d_in[8];
    const float* beta  = (const float*)d_in[9];
    float* out = (float*)d_out;

    __bf16* wsb = (__bf16*)d_ws;
    const size_t WSZ = (size_t)512 * 512;            // weight elements
    const size_t TSZ = (size_t)B_ * S_ * D_;         // tensor elements

    __bf16* Wq_bf = wsb;
    __bf16* Wk_bf = wsb + WSZ;
    __bf16* Wv_bf = wsb + 2 * WSZ;
    __bf16* Wm_bf = wsb + 3 * WSZ;
    __bf16* qb    = wsb + 4 * WSZ;
    __bf16* kb    = qb + TSZ;
    __bf16* vTb   = kb + TSZ;
    __bf16* ctxb  = vTb + TSZ;

    prep_kernel<<<1024, 256, 0, stream>>>(Wq, Wk, Wv, Wm, Wq_bf, Wk_bf, Wv_bf, Wm_bf);
    qkv_rope_kernel<<<dim3(64, 3, B_), 256, 0, stream>>>(
        x, Wq_bf, Wk_bf, Wv_bf, bq, bk, qb, kb, vTb);
    attn_kernel<<<dim3(64, H_, B_), 32, 0, stream>>>(qb, kb, vTb, ctxb);
    proj_ln_kernel<<<dim3(64, B_), 256, 0, stream>>>(
        ctxb, Wm_bf, bm, x, gamma, beta, out);
}
